// GRUModel_33303176413822
// MI455X (gfx1250) — matrix operand order repaired and checked by reading, not compiled, not run on hardware
//
#include <hip/hip_runtime.h>

typedef unsigned short u16;
typedef __attribute__((ext_vector_type(16))) __bf16 v16bf;
typedef __attribute__((ext_vector_type(8)))  __bf16 v8bf;
typedef __attribute__((ext_vector_type(8)))  float   v8f;

// ---------- bf16 helpers ----------
__device__ inline u16 f2bf(float f) {
  unsigned u = __float_as_uint(f);
  u += 0x7fffu + ((u >> 16) & 1u);           // round-to-nearest-even
  return (u16)(u >> 16);
}
__device__ inline float bf2f(u16 h) { return __uint_as_float(((unsigned)h) << 16); }

__device__ inline v16bf combine16(v8bf lo, v8bf hi) {
  v16bf r;
#pragma unroll
  for (int i = 0; i < 8; ++i) { r[i] = lo[i]; r[i + 8] = hi[i]; }
  return r;
}

// A fragment (16x32 bf16, MxK) from row-major [rows x stride] at (m0,k0).
// Lane L holds row m0+(L&15), contiguous K range half*16..half*16+15.
__device__ inline v16bf load_a(const u16* base, int stride, int m0, int k0, int lane) {
  int row  = m0 + (lane & 15);
  int half = lane >> 4;
  const u16* p = base + (long)row * stride + k0 + half * 16;
  v8bf lo = *(const v8bf*)(p);
  v8bf hi = *(const v8bf*)(p + 8);
  return combine16(lo, hi);
}

// B fragment (32x16 bf16, KxN): B(k,n) = W[n0+n][k0+k], W row-major [N x stride].
// Lane L holds column n0+(L&15), contiguous K range half*16..half*16+15.
__device__ inline v16bf load_b(const u16* W, int stride, int n0, int k0, int lane) {
  int col  = n0 + (lane & 15);
  int half = lane >> 4;
  const u16* p = W + (long)col * stride + k0 + half * 16;
  v8bf lo = *(const v8bf*)(p);
  v8bf hi = *(const v8bf*)(p + 8);
  return combine16(lo, hi);
}

#define LDSW_STRIDE 1032  // 1024 + 8 halves pad: 516 dwords, 516%64=4 -> banks spread

// B fragment from LDS-resident W tile (48 rows x LDSW_STRIDE halves).
__device__ inline v16bf load_b_lds(const u16* sW, int gate, int k0, int lane) {
  int rl   = gate * 16 + (lane & 15);
  int half = lane >> 4;
  const u16* p = sW + rl * LDSW_STRIDE + k0 + half * 16;
  v8bf lo = *(const v8bf*)(p);
  v8bf hi = *(const v8bf*)(p + 8);
  return combine16(lo, hi);
}

__device__ inline v8f wmma_bf16(v16bf a, v16bf b, v8f c) {
  return __builtin_amdgcn_wmma_f32_16x16x32_bf16(false, a, false, b, (short)0, c,
                                                 false, false);
}

// ---------- device-wide sense-reversal barrier ----------
struct GBar { unsigned count; unsigned sense; };

__global__ void k_bar_init(GBar* bar) {
  if (threadIdx.x == 0 && blockIdx.x == 0) { bar->count = 0u; bar->sense = 0u; }
}

__device__ inline void grid_sync(GBar* bar, unsigned nwg, unsigned* phase) {
  __syncthreads();
  if (threadIdx.x == 0) {
    unsigned target = *phase + 1u;
    __threadfence();
    unsigned arrived = __hip_atomic_fetch_add(&bar->count, 1u, __ATOMIC_ACQ_REL,
                                              __HIP_MEMORY_SCOPE_AGENT);
    if (arrived == nwg - 1u) {
      __hip_atomic_store(&bar->count, 0u, __ATOMIC_RELAXED, __HIP_MEMORY_SCOPE_AGENT);
      __hip_atomic_store(&bar->sense, target, __ATOMIC_RELEASE, __HIP_MEMORY_SCOPE_AGENT);
    } else {
      while (__hip_atomic_load(&bar->sense, __ATOMIC_ACQUIRE,
                               __HIP_MEMORY_SCOPE_AGENT) < target)
        __builtin_amdgcn_s_sleep(2);
    }
    *phase = target;
  }
  __syncthreads();
}

// ---------- kernels ----------
__global__ void k_cvt_bf16(const float* __restrict__ in, u16* __restrict__ out, int n) {
  int i = blockIdx.x * blockDim.x + threadIdx.x;
  if (i < n) out[i] = f2bf(in[i]);
}

// Layer-0 input gates: K=8 (trivially memory bound), xg = x @ W_ih0^T + b_ih0
__global__ void k_xg0(const float* __restrict__ x, const float* __restrict__ w,
                      const float* __restrict__ b, u16* __restrict__ xg) {
  long i = (long)blockIdx.x * blockDim.x + threadIdx.x;
  if (i >= (long)16384 * 3072) return;
  int g  = (int)(i % 3072);
  int bt = (int)(i / 3072);
  float acc = b[g];
  const float* xr = x + (long)bt * 8;
  const float* wr = w + (long)g * 8;
#pragma unroll
  for (int d = 0; d < 8; ++d) acc += xr[d] * wr[d];
  xg[i] = f2bf(acc);
}

// Input-gate GEMM for layers 1/2: Out[16384x3072] = A[16384x1024] @ W^T + bias (bf16 WMMA)
__global__ __launch_bounds__(256) void k_gemm_xg(const u16* __restrict__ A,
                                                 const u16* __restrict__ W,
                                                 const float* __restrict__ bias,
                                                 u16* __restrict__ Out) {
  const int K = 1024, N = 3072;
  int lane = threadIdx.x & 31;
  int wave = threadIdx.x >> 5;                 // 8 waves: 4 (M) x 2 (N)
  int m0 = blockIdx.y * 128 + (wave & 3) * 32;
  int n0 = blockIdx.x * 64  + (wave >> 2) * 32;
  int nn = lane & 15;
  float b0 = bias[n0 + nn], b1 = bias[n0 + 16 + nn];
  v8f c00, c01, c10, c11;
#pragma unroll
  for (int i = 0; i < 8; ++i) { c00[i] = b0; c01[i] = b1; c10[i] = b0; c11[i] = b1; }
  for (int k0 = 0; k0 < K; k0 += 32) {
    v16bf a0  = load_a(A, K, m0,      k0, lane);
    v16bf a1  = load_a(A, K, m0 + 16, k0, lane);
    v16bf bb0 = load_b(W, K, n0,      k0, lane);
    v16bf bb1 = load_b(W, K, n0 + 16, k0, lane);
    c00 = wmma_bf16(a0, bb0, c00);
    c01 = wmma_bf16(a0, bb1, c01);
    c10 = wmma_bf16(a1, bb0, c10);
    c11 = wmma_bf16(a1, bb1, c11);
  }
  int hi = lane >> 4;
#pragma unroll
  for (int i = 0; i < 8; ++i) {
    long r0 = m0 + i + 8 * hi, r1 = m0 + 16 + i + 8 * hi;
    Out[r0 * N + n0 + nn]      = f2bf(c00[i]);
    Out[r0 * N + n0 + 16 + nn] = f2bf(c01[i]);
    Out[r1 * N + n0 + nn]      = f2bf(c10[i]);
    Out[r1 * N + n0 + 16 + nn] = f2bf(c11[i]);
  }
}

// Persistent GRU recurrence: 64 WGs x 64 threads; WG owns 16 hidden units (3 gates).
// W_hh tile (48x1024 bf16, ~97KB) is LDS-resident for all 512 steps; h ping-pongs
// through L2-resident global buffers with one device-wide sync per timestep.
__global__ __launch_bounds__(64) void k_gru_rec(const u16* __restrict__ xg,
                                                const u16* __restrict__ Whh,
                                                const float* __restrict__ bhh,
                                                float* __restrict__ hM,
                                                u16* __restrict__ hb0,
                                                u16* __restrict__ hb1,
                                                u16* __restrict__ y,
                                                GBar* bar) {
  const int H = 1024, T = 512, G3 = 3072;
  const unsigned NWG = 64;
  __shared__ u16 sW[48 * LDSW_STRIDE];         // 99072 B
  unsigned phase = 0;
  int lane = threadIdx.x & 31;
  int wave = threadIdx.x >> 5;     // 0..1 -> batch rows [0,16) / [16,32)
  int j0 = blockIdx.x * 16;
  int nn = lane & 15, hi = lane >> 4;
  int m0 = wave * 16;

  // Stage our 48-row W_hh tile into LDS (once per layer): rows rl = gate*16 + unit.
  for (int c = threadIdx.x; c < 48 * 128; c += blockDim.x) {
    int rl  = c >> 7;                 // 0..47
    int off = (c & 127) * 8;          // halves within row, 16B chunks
    int wrow = (rl >> 4) * H + j0 + (rl & 15);
    v8bf v = *(const v8bf*)(Whh + (long)wrow * H + off);
    *(v8bf*)(sW + rl * LDSW_STRIDE + off) = v;
  }

  // zero our column slice of the state buffers
  for (int r = threadIdx.x; r < 32 * 16; r += blockDim.x) {
    int row = r >> 4, col = j0 + (r & 15);
    hM[row * H + col]  = 0.f;
    hb0[row * H + col] = 0;
    hb1[row * H + col] = 0;
  }
  grid_sync(bar, NWG, &phase);       // includes __syncthreads for the LDS staging

  int jg = j0 + nn;
  float br = bhh[jg], bz = bhh[H + jg], bn = bhh[2 * H + jg];

  for (int t = 0; t < T; ++t) {
    const u16* hin  = (t & 1) ? hb1 : hb0;
    u16*       hout = (t & 1) ? hb0 : hb1;

    // prefetch next step's input gates while this step's WMMA chain runs
    if (t + 1 < T) {
      long nb = ((long)(m0 + nn) * T + (t + 1)) * G3 + j0;
      __builtin_prefetch(xg + nb, 0, 1);
      __builtin_prefetch(xg + nb + H, 0, 1);
      __builtin_prefetch(xg + nb + 2 * H, 0, 1);
    }

    v8f cr, cz, cn;
#pragma unroll
    for (int i = 0; i < 8; ++i) { cr[i] = br; cz[i] = bz; cn[i] = bn; }
    for (int k0 = 0; k0 < H; k0 += 32) {
      v16bf a  = load_a(hin, H, m0, k0, lane);
      v16bf wr = load_b_lds(sW, 0, k0, lane);
      v16bf wz = load_b_lds(sW, 1, k0, lane);
      v16bf wn = load_b_lds(sW, 2, k0, lane);
      cr = wmma_bf16(a, wr, cr);
      cz = wmma_bf16(a, wz, cz);
      cn = wmma_bf16(a, wn, cn);
    }
#pragma unroll
    for (int i = 0; i < 8; ++i) {
      int row = m0 + i + 8 * hi;                   // batch 0..31
      long xbase = ((long)row * T + t) * G3;
      float xr = bf2f(xg[xbase + jg]);
      float xz = bf2f(xg[xbase + H + jg]);
      float xn = bf2f(xg[xbase + 2 * H + jg]);
      float rg = 1.f / (1.f + __expf(-(xr + cr[i])));
      float zg = 1.f / (1.f + __expf(-(xz + cz[i])));
      float ng = tanhf(xn + rg * cn[i]);
      float hprev = hM[row * H + jg];
      float hnew  = (1.f - zg) * ng + zg * hprev;
      hM[row * H + jg] = hnew;
      u16 hb = f2bf(hnew);
      hout[row * H + jg] = hb;
      y[((long)row * T + t) * H + jg] = hb;
    }
    grid_sync(bar, NWG, &phase);
  }
}

__global__ void k_meanpool(const u16* __restrict__ y, float* __restrict__ out) {
  int i = blockIdx.x * blockDim.x + threadIdx.x;
  if (i >= 32 * 1024) return;
  int b = i >> 10, j = i & 1023;
  float s = 0.f;
  const u16* p = y + (long)b * 512 * 1024 + j;
  for (int t = 0; t < 512; ++t) s += bf2f(p[(long)t * 1024]);
  out[i] = s * (1.f / 512.f);
}

__global__ void k_fc(const float* __restrict__ in, const float* __restrict__ w,
                     const float* __restrict__ b, float* __restrict__ out,
                     int Bsz, int K, int N, int relu) {
  int i = blockIdx.x * blockDim.x + threadIdx.x;
  if (i >= Bsz * N) return;
  int bi = i / N, o = i % N;
  float acc = b[o];
  const float* x  = in + (long)bi * K;
  const float* ww = w  + (long)o * K;
  for (int k = 0; k < K; ++k) acc += x[k] * ww[k];
  out[i] = relu ? fmaxf(acc, 0.f) : acc;
}

// ---------- host ----------
extern "C" void kernel_launch(void* const* d_in, const int* in_sizes, int n_in,
                              void* d_out, int out_size, void* d_ws, size_t ws_size,
                              hipStream_t stream) {
  (void)in_sizes; (void)n_in; (void)out_size; (void)ws_size;
  const float* x      = (const float*)d_in[0];
  const float* W_ih[3] = {(const float*)d_in[1], (const float*)d_in[5], (const float*)d_in[9]};
  const float* W_hh[3] = {(const float*)d_in[2], (const float*)d_in[6], (const float*)d_in[10]};
  const float* b_ih[3] = {(const float*)d_in[3], (const float*)d_in[7], (const float*)d_in[11]};
  const float* b_hh[3] = {(const float*)d_in[4], (const float*)d_in[8], (const float*)d_in[12]};
  const float* fc1w = (const float*)d_in[13];
  const float* fc1b = (const float*)d_in[14];
  const float* fc2w = (const float*)d_in[15];
  const float* fc2b = (const float*)d_in[16];
  const float* fc3w = (const float*)d_in[17];
  const float* fc3b = (const float*)d_in[18];

  char* ws = (char*)d_ws;
  size_t off = 0;
  auto alloc = [&](size_t bytes) -> char* {
    off = (off + 255) & ~(size_t)255;
    char* p = ws + off;
    off += bytes;
    return p;
  };

  const size_t WN = (size_t)3072 * 1024;          // weight elements per matrix
  u16* whh_bf[3]; u16* wih_bf[2];
  for (int l = 0; l < 3; ++l) whh_bf[l] = (u16*)alloc(WN * 2);
  for (int l = 0; l < 2; ++l) wih_bf[l] = (u16*)alloc(WN * 2);
  u16*   xg  = (u16*)alloc((size_t)16384 * 3072 * 2);
  float* hM  = (float*)alloc((size_t)32 * 1024 * 4);
  u16*   hb0 = (u16*)alloc((size_t)32 * 1024 * 2);
  u16*   hb1 = (u16*)alloc((size_t)32 * 1024 * 2);
  u16*   yA  = (u16*)alloc((size_t)32 * 512 * 1024 * 2);
  u16*   yB  = (u16*)alloc((size_t)32 * 512 * 1024 * 2);
  float* pooled = (float*)alloc((size_t)32 * 1024 * 4);
  float* fc1o   = (float*)alloc((size_t)32 * 512 * 4);
  float* fc2o   = (float*)alloc((size_t)32 * 256 * 4);
  GBar*  bar    = (GBar*)alloc(256);

  const int cvtB = (int)((WN + 255) / 256);
  for (int l = 0; l < 3; ++l)
    k_cvt_bf16<<<cvtB, 256, 0, stream>>>(W_hh[l], whh_bf[l], (int)WN);
  for (int l = 0; l < 2; ++l)
    k_cvt_bf16<<<cvtB, 256, 0, stream>>>(W_ih[l + 1], wih_bf[l], (int)WN);

  // layer 0
  k_xg0<<<(int)(((long)16384 * 3072 + 255) / 256), 256, 0, stream>>>(x, W_ih[0], b_ih[0], xg);
  k_bar_init<<<1, 64, 0, stream>>>(bar);
  k_gru_rec<<<64, 64, 0, stream>>>(xg, whh_bf[0], b_hh[0], hM, hb0, hb1, yA, bar);

  // layer 1
  {
    dim3 g(48, 128);
    k_gemm_xg<<<g, 256, 0, stream>>>(yA, wih_bf[0], b_ih[1], xg);
  }
  k_bar_init<<<1, 64, 0, stream>>>(bar);
  k_gru_rec<<<64, 64, 0, stream>>>(xg, whh_bf[1], b_hh[1], hM, hb0, hb1, yB, bar);

  // layer 2
  {
    dim3 g(48, 128);
    k_gemm_xg<<<g, 256, 0, stream>>>(yB, wih_bf[1], b_ih[2], xg);
  }
  k_bar_init<<<1, 64, 0, stream>>>(bar);
  k_gru_rec<<<64, 64, 0, stream>>>(xg, whh_bf[2], b_hh[2], hM, hb0, hb1, yA, bar);

  // head
  k_meanpool<<<(32 * 1024 + 255) / 256, 256, 0, stream>>>(yA, pooled);
  k_fc<<<(32 * 512 + 255) / 256, 256, 0, stream>>>(pooled, fc1w, fc1b, fc1o, 32, 1024, 512, 1);
  k_fc<<<(32 * 256 + 255) / 256, 256, 0, stream>>>(fc1o, fc2w, fc2b, fc2o, 32, 512, 256, 1);
  k_fc<<<1, 32, 0, stream>>>(fc2o, fc3w, fc3b, (float*)d_out, 32, 256, 1, 0);
}
